// RAttention_15470472200815
// MI455X (gfx1250) — matrix operand order repaired and checked by reading, not compiled, not run on hardware
//
#include <hip/hip_runtime.h>

// ---------------------------------------------------------------------------
// RAttention (criss-cross attention) for MI455X / gfx1250, wave32 + WMMA f16.
// softmax over a size-1 axis == 1  =>  out_LR/out_RL are all-ones and
// out_sum = out_H + out_W + 2.  All GEMMs use v_wmma_f32_16x16x32_f16.
// Each wave carries 4 accumulators and reuses its most expensive operand
// fragment across them (4x traffic cut on the memory-bound path).
// ---------------------------------------------------------------------------

constexpr int kB  = 4;
constexpr int kC  = 512;
constexpr int kC8 = 64;
constexpr int kH  = 128;
constexpr int kW  = 128;
constexpr int kHW = kH * kW;          // 16384 pixels per batch image

typedef __attribute__((ext_vector_type(16))) _Float16 v16h;
typedef __attribute__((ext_vector_type(8)))  _Float16 v8h;
typedef __attribute__((ext_vector_type(8)))  float    v8f;

__device__ __forceinline__ v8f wmma_f16(v16h a, v16h b, v8f c) {
  // D = A(16x32 f16) * B(32x16 f16) + C(16x16 f32)
  return __builtin_amdgcn_wmma_f32_16x16x32_f16(
      /*neg_a=*/false, a, /*neg_b=*/false, b,
      /*c_mod=*/(short)0, c, /*reuse_a=*/false, /*reuse_b=*/false);
}

// A-fragment layout (16x32 f16): lane L -> row M=L&15; elems 0..7 = K in
// [hi*8, hi*8+8), elems 8..15 = +16 (hi = L>=16).
// B-fragment layout (32x16 f16): lane L -> column N=L&15; elems follow A's K order.
// C/D layout: VGPR r, lane L -> (M = r + 8*hi, N = L&15).

// ---------------------------------------------------------------------------
// Kernel 1: projection GEMM  Y[b,o,p] = f16( sum_c W[o,c]*X[b,c,p] + bias[o] )
// One wave: 64(M=o) x 16(N=p) strip = 4 accumulators sharing one B fragment.
// ---------------------------------------------------------------------------
__global__ __launch_bounds__(256) void proj_kernel(
    const float* __restrict__ X,     // (kB, kC, kHW)
    const float* __restrict__ Wm,    // (O, kC)
    const float* __restrict__ bias,  // (O)
    _Float16*    __restrict__ Y,     // (kB, O, kHW) f16
    int O, int total_tiles)
{
  const int lane = threadIdx.x & 31;
  const int tile = blockIdx.x * 8 + (threadIdx.x >> 5);
  if (tile >= total_tiles) return;                 // wave-uniform
  const int MG  = O / 64;                          // groups of 4 m-tiles
  const int nt  = tile & 1023;                     // kHW/16 = 1024 pixel tiles
  const int rest = tile >> 10;
  const int mg  = rest % MG;
  const int b   = rest / MG;
  const int row = lane & 15;
  const int hi  = lane >> 4;

  const float* Bbase = X + (size_t)b * kC * kHW + (size_t)nt * 16;

  v8f acc0 = {}, acc1 = {}, acc2 = {}, acc3 = {};
  for (int k0 = 0; k0 < kC; k0 += 32) {
    // B fragment once: lane holds pixel column N = lane&15, K in A's order
    const float* bp = Bbase + (size_t)(k0 + hi * 8) * kHW + row;
    v16h bb;
    #pragma unroll
    for (int i = 0; i < 8; ++i) {
      bb[i]     = (_Float16)bp[(size_t)i * kHW];
      bb[8 + i] = (_Float16)bp[(size_t)(i + 16) * kHW];
    }

    #pragma unroll
    for (int t = 0; t < 4; ++t) {
      const int mrow = (mg * 4 + t) * 16 + row;
      const float* ap = Wm + (size_t)mrow * kC + hi * 8 + k0;
      float4 f0 = *(const float4*)(ap + 0);
      float4 f1 = *(const float4*)(ap + 4);
      float4 f2 = *(const float4*)(ap + 16);
      float4 f3 = *(const float4*)(ap + 20);
      v16h a;
      a[0]=(_Float16)f0.x;  a[1]=(_Float16)f0.y;  a[2]=(_Float16)f0.z;  a[3]=(_Float16)f0.w;
      a[4]=(_Float16)f1.x;  a[5]=(_Float16)f1.y;  a[6]=(_Float16)f1.z;  a[7]=(_Float16)f1.w;
      a[8]=(_Float16)f2.x;  a[9]=(_Float16)f2.y;  a[10]=(_Float16)f2.z; a[11]=(_Float16)f2.w;
      a[12]=(_Float16)f3.x; a[13]=(_Float16)f3.y; a[14]=(_Float16)f3.z; a[15]=(_Float16)f3.w;
      if      (t == 0) acc0 = wmma_f16(a, bb, acc0);
      else if (t == 1) acc1 = wmma_f16(a, bb, acc1);
      else if (t == 2) acc2 = wmma_f16(a, bb, acc2);
      else             acc3 = wmma_f16(a, bb, acc3);
    }
  }
  #pragma unroll
  for (int t = 0; t < 4; ++t) {
    const v8f acc = (t == 0) ? acc0 : (t == 1) ? acc1 : (t == 2) ? acc2 : acc3;
    #pragma unroll
    for (int r = 0; r < 8; ++r) {
      const int mr = (mg * 4 + t) * 16 + r + hi * 8;
      const float val = acc[r] + bias[mr];
      Y[((size_t)b * O + mr) * kHW + (size_t)nt * 16 + row] = (_Float16)val;
    }
  }
}

// ---------------------------------------------------------------------------
// Kernel 2: energies.  PART=0: energy_H per (b, w=s): S[h,j] = q[:,h,w].k[:,j,w]
//                      PART=1: energy_W per (b, h=s): S[w,j] = q[:,h,w].k[:,h,j]
// pixel(i) = base + i*stride.  One wave: 16(m) x 64(j) = 4 accumulators
// sharing one A gather (channel-strided q).  Diagonal of energy_H -> -inf.
// E layout: (kB, kHW, 256) f32.
// ---------------------------------------------------------------------------
template<int PART>
__global__ __launch_bounds__(256) void energy_kernel(
    const _Float16* __restrict__ qh, const _Float16* __restrict__ kh,
    float* __restrict__ E)
{
  const int lane = threadIdx.x & 31;
  const int tile = blockIdx.x * 8 + (threadIdx.x >> 5);  // 8192 tiles exactly
  const int ng = tile & 1;            // group of 4 n-tiles
  const int mt = (tile >> 1) & 7;
  const int s  = (tile >> 4) & 127;
  const int b  = tile >> 11;
  constexpr int stride = (PART == 0) ? kW : 1;
  const int base = (PART == 0) ? s : s * kW;
  const int row = lane & 15, hi = lane >> 4;
  const int m = mt * 16 + row;
  const size_t pixm = (size_t)base + (size_t)m * stride;
  const _Float16* Q = qh + (size_t)b * kC8 * kHW + pixm;
  const _Float16* Kb = kh + (size_t)b * kC8 * kHW + base;

  v8f acc0 = {}, acc1 = {}, acc2 = {}, acc3 = {};
  #pragma unroll
  for (int k0 = 0; k0 < kC8; k0 += 32) {
    v16h a;                                  // q gather, shared by 4 n-tiles
    #pragma unroll
    for (int i = 0; i < 8; ++i) {
      const int c0 = k0 + hi * 8 + i;
      a[i]     = Q[(size_t)c0 * kHW];
      a[8 + i] = Q[(size_t)(c0 + 16) * kHW];
    }
    const _Float16* Krow = Kb + (size_t)(k0 + hi * 8) * kHW;
    #pragma unroll
    for (int t = 0; t < 4; ++t) {
      const int nt = ng * 4 + t;
      const _Float16* Kp = Krow + (size_t)(nt * 16 + row) * stride;
      v16h bf;
      #pragma unroll
      for (int i = 0; i < 8; ++i) {
        bf[i]     = Kp[(size_t)i * kHW];
        bf[8 + i] = Kp[(size_t)(i + 16) * kHW];
      }
      if      (t == 0) acc0 = wmma_f16(a, bf, acc0);
      else if (t == 1) acc1 = wmma_f16(a, bf, acc1);
      else if (t == 2) acc2 = wmma_f16(a, bf, acc2);
      else             acc3 = wmma_f16(a, bf, acc3);
    }
  }
  constexpr int joff = (PART == 0) ? 0 : 128;
  #pragma unroll
  for (int t = 0; t < 4; ++t) {
    const v8f acc = (t == 0) ? acc0 : (t == 1) ? acc1 : (t == 2) ? acc2 : acc3;
    const int n = (ng * 4 + t) * 16 + row;
    #pragma unroll
    for (int r = 0; r < 8; ++r) {
      const int mr = mt * 16 + r + hi * 8;
      float val = acc[r];
      if (PART == 0 && mr == n) val = -__builtin_inff();  // diag mask (j == h)
      const size_t pm = (size_t)base + (size_t)mr * stride;
      E[((size_t)b * kHW + pm) * 256 + joff + n] = val;
    }
  }
}

// ---------------------------------------------------------------------------
// Kernel 3: softmax over the 256 concatenated logits, one wave per pixel.
// ---------------------------------------------------------------------------
__global__ __launch_bounds__(256) void softmax_kernel(
    const float* __restrict__ E, _Float16* __restrict__ att)
{
  const int lane = threadIdx.x & 31;
  const size_t p = (size_t)blockIdx.x * 8 + (threadIdx.x >> 5);
  const float* rp = E + p * 256 + lane * 8;
  float4 x0 = *(const float4*)rp;
  float4 x1 = *(const float4*)(rp + 4);
  float xs[8] = {x0.x, x0.y, x0.z, x0.w, x1.x, x1.y, x1.z, x1.w};
  float mx = xs[0];
  #pragma unroll
  for (int i = 1; i < 8; ++i) mx = fmaxf(mx, xs[i]);
  #pragma unroll
  for (int off = 16; off > 0; off >>= 1) mx = fmaxf(mx, __shfl_xor(mx, off, 32));
  float sum = 0.f;
  #pragma unroll
  for (int i = 0; i < 8; ++i) { xs[i] = __expf(xs[i] - mx); sum += xs[i]; }
  #pragma unroll
  for (int off = 16; off > 0; off >>= 1) sum += __shfl_xor(sum, off, 32);
  const float inv = 1.0f / sum;
  v8h o;
  #pragma unroll
  for (int i = 0; i < 8; ++i) o[i] = (_Float16)(xs[i] * inv);
  *(v8h*)(att + p * 256 + lane * 8) = o;
}

// ---------------------------------------------------------------------------
// Kernel 4: out_H / out_W.  D[m, c] = sum_j att[pix(m), joff+j] * v[c, pix(j)]
// PART=0 (out_H, fixed w=s): pix(i)=i*kW+s ; PART=1 (out_W, fixed h=s): pix=s*kW+i.
// One wave: 64(m pixels) x 16(c) = 4 accumulators sharing one B gather
// (channel-strided v).  ACCUM=false writes D + 2, ACCUM=true accumulates.
// ---------------------------------------------------------------------------
template<int PART, bool ACCUM>
__global__ __launch_bounds__(256) void out_kernel(
    const _Float16* __restrict__ att, const _Float16* __restrict__ vh,
    float* __restrict__ osum)
{
  const int lane = threadIdx.x & 31;
  const int tile = blockIdx.x * 8 + (threadIdx.x >> 5);  // 32768 tiles exactly
  const int ct = tile & 31;           // channel tile (512/16)
  const int mg = (tile >> 5) & 1;     // group of 4 pixel tiles
  const int s  = (tile >> 6) & 127;
  const int b  = tile >> 13;
  constexpr int stride = (PART == 0) ? kW : 1;
  constexpr int joff   = (PART == 0) ? 0 : 128;
  const int base = (PART == 0) ? s : s * kW;
  const int row = lane & 15, hi = lane >> 4;
  const _Float16* Abase = att + ((size_t)b * kHW) * 256 + joff;
  const _Float16* Vb    = vh + ((size_t)b * kC + ct * 16) * kHW + base;

  v8f acc0 = {}, acc1 = {}, acc2 = {}, acc3 = {};
  #pragma unroll
  for (int k0 = 0; k0 < kH; k0 += 32) {
    const int j = k0 + hi * 8;
    const _Float16* vp = Vb + (size_t)row * kHW + (size_t)j * stride;
    v16h bf;                                 // v gather, shared by 4 m-tiles
    #pragma unroll
    for (int i = 0; i < 8; ++i) { bf[i] = vp[(size_t)i * stride]; bf[8 + i] = vp[(size_t)(i + 16) * stride]; }
    #pragma unroll
    for (int t = 0; t < 4; ++t) {
      const int m = (mg * 4 + t) * 16 + row;
      const size_t pixm = (size_t)base + (size_t)m * stride;
      const _Float16* ap = Abase + pixm * 256 + k0 + hi * 8;
      v8h a0 = *(const v8h*)ap;
      v8h a1 = *(const v8h*)(ap + 16);
      v16h a;
      #pragma unroll
      for (int i = 0; i < 8; ++i) { a[i] = a0[i]; a[8 + i] = a1[i]; }
      if      (t == 0) acc0 = wmma_f16(a, bf, acc0);
      else if (t == 1) acc1 = wmma_f16(a, bf, acc1);
      else if (t == 2) acc2 = wmma_f16(a, bf, acc2);
      else             acc3 = wmma_f16(a, bf, acc3);
    }
  }
  const int c = ct * 16 + row;
  #pragma unroll
  for (int t = 0; t < 4; ++t) {
    const v8f acc = (t == 0) ? acc0 : (t == 1) ? acc1 : (t == 2) ? acc2 : acc3;
    #pragma unroll
    for (int r = 0; r < 8; ++r) {
      const int mr = (mg * 4 + t) * 16 + r + hi * 8;
      const size_t pm = (size_t)base + (size_t)mr * stride;
      const size_t idx = ((size_t)b * kC + c) * kHW + pm;
      if (ACCUM) osum[idx] += acc[r];
      else       osum[idx] = acc[r] + 2.0f;   // + out_LR + out_RL (== 1 each)
    }
  }
}

// ---------------------------------------------------------------------------
// Kernel 5: residual epilogue.  out1 = g1*out_sum + x_exmplar,
//                               out2 = g2*out_sum + x_query.  float4 streams.
// ---------------------------------------------------------------------------
__global__ __launch_bounds__(256) void final_kernel(
    const float* __restrict__ osum, const float* __restrict__ xe,
    const float* __restrict__ xq, const float* __restrict__ g1p,
    const float* __restrict__ g2p, float* __restrict__ o1,
    float* __restrict__ o2)
{
  const size_t i = ((size_t)blockIdx.x * 256 + threadIdx.x) * 4;
  const float g1 = g1p[0], g2 = g2p[0];
  float4 sv = *(const float4*)(osum + i);
  float4 ev = *(const float4*)(xe + i);
  float4 qv = *(const float4*)(xq + i);
  float4 r1, r2;
  r1.x = fmaf(g1, sv.x, ev.x); r1.y = fmaf(g1, sv.y, ev.y);
  r1.z = fmaf(g1, sv.z, ev.z); r1.w = fmaf(g1, sv.w, ev.w);
  r2.x = fmaf(g2, sv.x, qv.x); r2.y = fmaf(g2, sv.y, qv.y);
  r2.z = fmaf(g2, sv.z, qv.z); r2.w = fmaf(g2, sv.w, qv.w);
  *(float4*)(o1 + i) = r1;
  *(float4*)(o2 + i) = r2;
}

// ---------------------------------------------------------------------------
extern "C" void kernel_launch(void* const* d_in, const int* in_sizes, int n_in,
                              void* d_out, int out_size, void* d_ws, size_t ws_size,
                              hipStream_t stream) {
  (void)in_sizes; (void)n_in; (void)out_size; (void)ws_size;
  const float* x_ex = (const float*)d_in[0];
  const float* x_q  = (const float*)d_in[1];
  const float* Wq   = (const float*)d_in[2];
  const float* bq   = (const float*)d_in[3];
  const float* Wk   = (const float*)d_in[4];
  const float* bk   = (const float*)d_in[5];
  const float* Wv   = (const float*)d_in[6];
  const float* bv   = (const float*)d_in[7];
  const float* g1   = (const float*)d_in[8];
  const float* g2   = (const float*)d_in[9];

  const size_t QN = (size_t)kB * kC8 * kHW;   //  4 Mi halfs  (q / k)
  const size_t VN = (size_t)kB * kC  * kHW;   // 32 Mi elements (v / out_sum)
  const size_t EN = (size_t)kB * kHW * 256;   // 16 Mi elements (energies / att)

  // workspace carve-up (~304 MB total)
  _Float16* qh   = (_Float16*)d_ws;
  _Float16* kh   = qh + QN;
  _Float16* vh   = kh + QN;
  float*    E    = (float*)(vh + VN);
  _Float16* att  = (_Float16*)(E + EN);
  float*    osum = (float*)(att + EN);

  // 1) projections (WMMA GEMMs, 8 waves / block, 64x16 strip / wave)
  const int tiles_qk = kB * (kC8 / 64) * (kHW / 16);   //  4096
  const int tiles_v  = kB * (kC  / 64) * (kHW / 16);   // 32768
  proj_kernel<<<(tiles_qk + 7) / 8, 256, 0, stream>>>(x_q,  Wq, bq, qh, kC8, tiles_qk);
  proj_kernel<<<(tiles_qk + 7) / 8, 256, 0, stream>>>(x_ex, Wk, bk, kh, kC8, tiles_qk);
  proj_kernel<<<(tiles_v  + 7) / 8, 256, 0, stream>>>(x_ex, Wv, bv, vh, kC,  tiles_v);

  // 2) energies (H-part with diagonal mask, then W-part); 8192 tiles each
  energy_kernel<0><<<1024, 256, 0, stream>>>(qh, kh, E);
  energy_kernel<1><<<1024, 256, 0, stream>>>(qh, kh, E);

  // 3) joint softmax over 256 logits per pixel
  softmax_kernel<<<(kB * kHW) / 8, 256, 0, stream>>>(E, att);

  // 4) out_H (writes D+2), then out_W (accumulates); 32768 tiles each
  out_kernel<0, false><<<4096, 256, 0, stream>>>(att, vh, osum);
  out_kernel<1, true ><<<4096, 256, 0, stream>>>(att, vh, osum);

  // 5) residual epilogue -> two concatenated outputs
  float* out1 = (float*)d_out;
  float* out2 = out1 + VN;
  final_kernel<<<(int)(VN / 1024), 256, 0, stream>>>(osum, x_ex, x_q, g1, g2, out1, out2);
}
